// MessagePassingLayer_7499012899534
// MI455X (gfx1250) — matrix operand order repaired and checked by reading, not compiled, not run on hardware
//
#include <hip/hip_runtime.h>

// ---------------------------------------------------------------------------
// Fused message-passing layer for MI455X (gfx1250, wave32, WMMA).
// N=32768 nodes, K=32 neighbors, F=4, W=256.
// Dominant: two [1M x 256] x [256 x 256] GEMMs (274 GFLOP) -> bf16 WMMA,
// f32 accumulate. HBM floor ~1.15 GB (h_neighbors streamed once, coalesced,
// through LDS). Weights pre-converted to bf16 in d_ws (L2-resident).
// ---------------------------------------------------------------------------

#define N_NODES  32768
#define KNBR     32
#define FFEAT    4
#define WDIM     256
#define NODES_PER_BLK 4
#define EDGES_PER_BLK (NODES_PER_BLK * KNBR)   // 128
#define THREADS  256
#define LDS_STRIDE 264                          // 256 + 8 bf16 pad

typedef __attribute__((ext_vector_type(16))) __bf16 v16bf;
typedef __attribute__((ext_vector_type(8)))  __bf16 v8bf;
typedef __attribute__((ext_vector_type(8)))  float  v8f;
typedef __attribute__((ext_vector_type(2)))  float  v2f;

__device__ __forceinline__ float silu_f(float x) {
    // fast: v_exp_f32 + v_rcp_f32 (approx reciprocal, not the IEEE div chain)
    return x * __builtin_amdgcn_rcpf(1.0f + __expf(-x));
}

// A fragment (16x32 bf16) from LDS, CDNA5 layout:
// lanes 0-15: row = lane,    K in {k0..k0+7, k0+16..k0+23}
// lanes16-31: row = lane-16, K in {k0+8..k0+15, k0+24..k0+31}
__device__ __forceinline__ v16bf load_a_frag(const __bf16* lds, int row0, int k0, int lane) {
    const int row  = row0 + (lane & 15);
    const int koff = k0 + ((lane >> 4) << 3);
    const __bf16* p = lds + row * LDS_STRIDE + koff;
    union { v16bf v; v8bf h[2]; } u;
    u.h[0] = *(const v8bf*)(p);
    u.h[1] = *(const v8bf*)(p + 16);
    return u.v;
}

// B fragment (32x16) from pre-converted bf16 weights (L2-resident).
__device__ __forceinline__ v16bf load_b_frag_bf16(const __bf16* __restrict__ Wm,
                                                  int k0, int c0, int lane) {
    const __bf16* p = Wm + (k0 + ((lane >> 4) << 3)) * WDIM + c0 + (lane & 15);
    v16bf r;
#pragma unroll
    for (int i = 0; i < 16; ++i) r[i] = p[((i & 7) + ((i >> 3) << 4)) * WDIM];
    return r;
}

// Fallback: B fragment from fp32 weights, convert in-register.
__device__ __forceinline__ v16bf load_b_frag_f32(const float* __restrict__ Wm,
                                                 int k0, int c0, int lane) {
    const float* p = Wm + (long)(k0 + ((lane >> 4) << 3)) * WDIM + c0 + (lane & 15);
    v16bf r;
#pragma unroll
    for (int i = 0; i < 16; ++i) r[i] = (__bf16)p[((i & 7) + ((i >> 3) << 4)) * WDIM];
    return r;
}

// one-shot: fp32 weights -> bf16 in workspace
__global__ void convert_weights_kernel(const float* __restrict__ a,
                                       const float* __restrict__ b,
                                       __bf16* __restrict__ oa,
                                       __bf16* __restrict__ ob) {
    const int i = blockIdx.x * blockDim.x + threadIdx.x;   // 0..65535
    oa[i] = (__bf16)a[i];
    ob[i] = (__bf16)b[i];
}

template <bool BF16W>
__global__ __launch_bounds__(THREADS, 2)
void msg_pass_fused_kernel(const float* __restrict__ h_center,
                           const float* __restrict__ h_neighbors,
                           const float* __restrict__ differences,
                           const float* __restrict__ W_f1, const float* __restrict__ b_f1,
                           const void*  __restrict__ W_f2p, const float* __restrict__ b_f2,
                           const void*  __restrict__ W_nbp, const float* __restrict__ b_nb,
                           const float* __restrict__ W_c,  const float* __restrict__ b_c,
                           float* __restrict__ out)
{
    __shared__ __align__(16) __bf16 lds_f[EDGES_PER_BLK * LDS_STRIDE];  // 66 KB
    __shared__ __align__(16) __bf16 lds_h[EDGES_PER_BLK * LDS_STRIDE];  // 66 KB
    __shared__ __align__(16) float  lds_diff[EDGES_PER_BLK * FFEAT];    // 2 KB
    __shared__ __align__(16) float  lds_msg[NODES_PER_BLK * WDIM];      // 4 KB

    const int tid  = threadIdx.x;
    const int lane = tid & 31;
    const int wave = tid >> 5;                 // 0..7 -> edge row-tile id
    const long edge0 = (long)blockIdx.x * EDGES_PER_BLK;

    // ---- zero msg accumulator -------------------------------------------
#pragma unroll
    for (int i = 0; i < (NODES_PER_BLK * WDIM) / THREADS; ++i)
        lds_msg[tid + THREADS * i] = 0.0f;

    // ---- stage differences tile (128 x 4 f32) ---------------------------
    if (tid < EDGES_PER_BLK) {
        float4 d = *(const float4*)(differences + (edge0 + tid) * FFEAT);
        *(float4*)(lds_diff + tid * FFEAT) = d;
    }

    // ---- stage h_neighbors tile (128 x 256 f32 -> bf16 LDS) -------------
    // Fully coalesced: each iteration the block reads a contiguous 4 KB
    // stripe (each wave 512 B contiguous); dominant HBM stream of the layer.
    {
        const float* src = h_neighbors + edge0 * WDIM;
#pragma unroll 4
        for (int it = 0; it < (EDGES_PER_BLK * WDIM) / (THREADS * 4); ++it) {
            const int f = it * (THREADS * 4) + tid * 4;    // flat f32 index
            float4 q = *(const float4*)(src + f);
            const int row = f >> 8;                        // / WDIM
            const int col = f & (WDIM - 1);
            struct __align__(8) B4 { __bf16 v[4]; } b;
            b.v[0] = (__bf16)q.x; b.v[1] = (__bf16)q.y;
            b.v[2] = (__bf16)q.z; b.v[3] = (__bf16)q.w;
            *(B4*)(lds_h + row * LDS_STRIDE + col) = b;
        }
    }
    __syncthreads();

    // ---- GEMM1: f = silu(diff @ W_f1 + b_f1) via V_WMMA_F32_16X16X4_F32 -
    {
        const int r = wave;                       // row tile (16 edges)
        const int m = lane & 15;
        const int kh = lane >> 4;                 // 0/1 -> K pair select
        v2f a;
        a.x = lds_diff[(r * 16 + m) * FFEAT + kh * 2 + 0];
        a.y = lds_diff[(r * 16 + m) * FFEAT + kh * 2 + 1];
        for (int c = 0; c < 16; ++c) {
            const int col = c * 16 + m;
            v2f b;
            b.x = W_f1[(kh + 0) * WDIM + col];
            b.y = W_f1[(kh + 2) * WDIM + col];
            const float bias = b_f1[col];
            v8f acc;
#pragma unroll
            for (int i = 0; i < 8; ++i) acc[i] = bias;
            acc = __builtin_amdgcn_wmma_f32_16x16x4_f32(
                false, a, false, b, (short)0, acc, false, false);
#pragma unroll
            for (int v = 0; v < 8; ++v) {
                const int row = r * 16 + ((lane < 16) ? v : v + 8);
                lds_f[row * LDS_STRIDE + col] = (__bf16)silu_f(acc[v]);
            }
        }
    }
    __syncthreads();

    // ---- dual bf16 WMMA GEMMs + gate + neighbor reduction ---------------
    {
        const int r = wave;                       // 16-edge row tile
        const int m = lane & 15;
        const int node_local = r >> 1;            // 32 edges per node = 2 tiles

        // hoist all A fragments (8 K-steps x 2 matrices = 128 VGPRs),
        // reused across all 16 column tiles -> hot loop touches only B.
        v16bf aF[8], aH[8];
#pragma unroll
        for (int kt = 0; kt < 8; ++kt) {
            aF[kt] = load_a_frag(lds_f, r * 16, kt * 32, lane);
            aH[kt] = load_a_frag(lds_h, r * 16, kt * 32, lane);
        }

        for (int c = 0; c < 16; ++c) {
            const int col = c * 16 + m;
            const float bF = b_f2[col];
            const float bH = b_nb[col];
            v8f accF, accH;
#pragma unroll
            for (int i = 0; i < 8; ++i) { accF[i] = bF; accH[i] = bH; }
#pragma unroll
            for (int kt = 0; kt < 8; ++kt) {
                v16bf wF, wH;
                if constexpr (BF16W) {
                    wF = load_b_frag_bf16((const __bf16*)W_f2p, kt * 32, c * 16, lane);
                    wH = load_b_frag_bf16((const __bf16*)W_nbp, kt * 32, c * 16, lane);
                } else {
                    wF = load_b_frag_f32((const float*)W_f2p, kt * 32, c * 16, lane);
                    wH = load_b_frag_f32((const float*)W_nbp, kt * 32, c * 16, lane);
                }
                accF = __builtin_amdgcn_wmma_f32_16x16x32_bf16(
                    false, aF[kt], false, wF, (short)0, accF, false, false);
                accH = __builtin_amdgcn_wmma_f32_16x16x32_bf16(
                    false, aH[kt], false, wH, (short)0, accH, false, false);
            }
            // gated product; this lane holds 8 edge-rows of column `col`
            float part = 0.0f;
#pragma unroll
            for (int v = 0; v < 8; ++v) part += accF[v] * accH[v];
            // fold the two half-row lanes (L and L+16 share `col`)
            part += __shfl_xor(part, 16);
            if (lane < 16)
                atomicAdd(&lds_msg[node_local * WDIM + col], part);  // ds_add_f32
        }
    }
    __syncthreads();

    // ---- center path: h_out = silu(h_center @ W_c + b_c + msg) ----------
    {
        float* hcs = (float*)lds_f;               // reuse LDS (4 x 256 f32)
        float4 q = *(const float4*)(h_center +
                    (long)blockIdx.x * NODES_PER_BLK * WDIM + tid * 4);
        *(float4*)(hcs + tid * 4) = q;
        __syncthreads();

        const int nl = tid >> 6;                  // local node 0..3
        const int cg = (tid & 63) * 4;            // 4 contiguous columns
        float a0 = 0.f, a1 = 0.f, a2 = 0.f, a3 = 0.f;
        for (int k = 0; k < WDIM; ++k) {
            const float hv = hcs[nl * WDIM + k];
            float4 wr = *(const float4*)(W_c + k * WDIM + cg);
            a0 = fmaf(hv, wr.x, a0);
            a1 = fmaf(hv, wr.y, a1);
            a2 = fmaf(hv, wr.z, a2);
            a3 = fmaf(hv, wr.w, a3);
        }
        const float4 bb = *(const float4*)(b_c + cg);
        const float4 mm = *(const float4*)(lds_msg + nl * WDIM + cg);
        float4 o;
        o.x = silu_f(a0 + bb.x + mm.x);
        o.y = silu_f(a1 + bb.y + mm.y);
        o.z = silu_f(a2 + bb.z + mm.z);
        o.w = silu_f(a3 + bb.w + mm.w);
        const long node = (long)blockIdx.x * NODES_PER_BLK + nl;
        *(float4*)(out + node * WDIM + cg) = o;
    }
}

extern "C" void kernel_launch(void* const* d_in, const int* in_sizes, int n_in,
                              void* d_out, int out_size, void* d_ws, size_t ws_size,
                              hipStream_t stream) {
    (void)in_sizes; (void)n_in; (void)out_size;
    const float* h_center    = (const float*)d_in[0];
    const float* h_neighbors = (const float*)d_in[1];
    const float* differences = (const float*)d_in[2];
    const float* W_f1 = (const float*)d_in[3];
    const float* b_f1 = (const float*)d_in[4];
    const float* W_f2 = (const float*)d_in[5];
    const float* b_f2 = (const float*)d_in[6];
    const float* W_nb = (const float*)d_in[7];
    const float* b_nb = (const float*)d_in[8];
    const float* W_c  = (const float*)d_in[9];
    const float* b_c  = (const float*)d_in[10];
    float* out = (float*)d_out;

    dim3 grid(N_NODES / NODES_PER_BLK);   // 8192 blocks
    dim3 block(THREADS);                  // 8 wave32

    const size_t wbytes = (size_t)WDIM * WDIM * sizeof(__bf16);  // 128 KB each
    if (ws_size >= 2 * wbytes) {
        __bf16* wf2_bf = (__bf16*)d_ws;
        __bf16* wnb_bf = wf2_bf + WDIM * WDIM;
        convert_weights_kernel<<<dim3((WDIM * WDIM) / THREADS), dim3(THREADS), 0, stream>>>(
            W_f2, W_nb, wf2_bf, wnb_bf);
        msg_pass_fused_kernel<true><<<grid, block, 0, stream>>>(
            h_center, h_neighbors, differences,
            W_f1, b_f1, wf2_bf, b_f2, wnb_bf, b_nb, W_c, b_c, out);
    } else {
        msg_pass_fused_kernel<false><<<grid, block, 0, stream>>>(
            h_center, h_neighbors, differences,
            W_f1, b_f1, W_f2, b_f2, W_nb, b_nb, W_c, b_c, out);
    }
}
